// Decoder_78365973283048
// MI455X (gfx1250) — matrix operand order repaired and checked by reading, not compiled, not run on hardware
//
#include <hip/hip_runtime.h>
#include <cstdint>

#define S_STEPS 512
#define BATCH   64
#define C_MEL   128
#define H_DIM   512
#define A_DIM   256
#define TP      544        // T + 2*PAD
#define CUMLEN  560        // T + 2*(PAD+CUM_PAD)
#define NWG     64
#define INV_SQRT2 0.70710678118654752f

typedef __bf16 bf16_t;
typedef bf16_t bf16x16 __attribute__((ext_vector_type(16)));
typedef float  f32x8   __attribute__((ext_vector_type(8)));

__device__ __forceinline__ float sigf(float x){ return 1.0f/(1.0f + __expf(-x)); }
__device__ __forceinline__ float geluf(float x){
  float x3 = x*x*x;
  return 0.5f*x*(1.0f + tanhf(0.7978845608028654f*(x + 0.044715f*x3)));
}

// ---- A fragment: fp32 row-major -> bf16 WMMA A layout (16x32, 16-bit) ----
// lanes 0-15: M=lane, K = k0+[0..7] and k0+[16..23]
// lanes16-31: M=lane-16, K = k0+[8..15] and k0+[24..31]
__device__ __forceinline__ bf16x16 load_a_frag(const float* __restrict__ A, int lda,
                                               int m0, int k0, int lane){
  const float* p = A + (size_t)(m0 + (lane & 15))*lda + k0 + ((lane < 16) ? 0 : 8);
  bf16x16 r;
  #pragma unroll
  for (int e=0;e<8;e++) r[e] = (bf16_t)p[e];
  #pragma unroll
  for (int e=0;e<8;e++) r[8+e] = (bf16_t)p[16+e];
  return r;
}

// ---- device-wide sense barrier for the persistent scan kernels ----
__device__ __forceinline__ void gsync(int* bar){
  __syncthreads();
  if (threadIdx.x == 0){
    int g = __hip_atomic_load(&bar[1], __ATOMIC_ACQUIRE, __HIP_MEMORY_SCOPE_AGENT);
    int a = __hip_atomic_fetch_add(&bar[0], 1, __ATOMIC_ACQ_REL, __HIP_MEMORY_SCOPE_AGENT);
    if (a == NWG-1){
      __hip_atomic_store(&bar[0], 0, __ATOMIC_RELAXED, __HIP_MEMORY_SCOPE_AGENT);
      __hip_atomic_fetch_add(&bar[1], 1, __ATOMIC_ACQ_REL, __HIP_MEMORY_SCOPE_AGENT);
    } else {
      while (__hip_atomic_load(&bar[1], __ATOMIC_ACQUIRE, __HIP_MEMORY_SCOPE_AGENT) == g)
        __builtin_amdgcn_s_sleep(2);
    }
  }
  __syncthreads();
}

// ---- pack fp32 weight W (Nw x Ksrc, row-major) into bf16 WMMA B-fragment tiles.
// Tile(k0,n0) at ((n0/16)*(Ktot/32) + (kOff+k0)/32)*512, element = lane*16 + j,
// lane<16 -> K=k0+j (j<8) / k0+8+j (j>=8), lane>=16 -> K=k0+8+j (j<8) / k0+16+j (j>=8), N = n0+(lane&15).
__global__ __launch_bounds__(256)
void pack_b_kernel(const float* __restrict__ W, bf16_t* __restrict__ dst,
                   int Nw, int Ksrc, int Ktot, int kOff){
  int tile = blockIdx.x;
  int nTk  = Ksrc >> 5;
  int kt   = tile % nTk;
  int ntile= tile / nTk;
  for (int e = threadIdx.x; e < 512; e += 256){
    int lane = e >> 4, j = e & 15;
    int n = ntile*16 + (lane & 15);
    int k = kt*32 + ((lane < 16) ? 0 : 8) + j + ((j < 8) ? 0 : 8);
    float v = (n < Nw) ? W[(size_t)n*Ksrc + k] : 0.0f;
    size_t di = ((size_t)ntile*(Ktot >> 5) + ((kOff >> 5) + kt))*512 + e;
    dst[di] = (bf16_t)v;
  }
}

// ---- big WMMA GEMM: 128x128 workgroup tile, 8 waves, each wave 2 m-tiles x 4 n-tiles.
// Requires M%128==0, Nlog%128==0 (all large feed-forward layers qualify).
__global__ __launch_bounds__(256)
void gemm_bf16_big_kernel(const float* __restrict__ A, const bf16_t* __restrict__ Bp,
                          const float* __restrict__ bias, float* __restrict__ C,
                          int Nlog, int K, int act){
  const int tid  = threadIdx.x;
  const int w    = tid >> 5, lane = tid & 31;
  const int mgi  = w & 3;              // 4 m-groups of 32 rows
  const int ngi  = w >> 2;             // 2 n-groups of 64 cols
  const int m0   = blockIdx.x*128 + mgi*32;
  const int n0   = blockIdx.y*128 + ngi*64;
  const int nTper = K >> 5;
  const bf16_t* bb0 = Bp + ((size_t)((n0 >> 4) + 0)*nTper)*512 + lane*16;
  const bf16_t* bb1 = Bp + ((size_t)((n0 >> 4) + 1)*nTper)*512 + lane*16;
  const bf16_t* bb2 = Bp + ((size_t)((n0 >> 4) + 2)*nTper)*512 + lane*16;
  const bf16_t* bb3 = Bp + ((size_t)((n0 >> 4) + 3)*nTper)*512 + lane*16;

  f32x8 acc[2][4] = {};
  #pragma unroll 2
  for (int kk = 0; kk < K; kk += 32){
    const size_t ko = (size_t)(kk >> 5)*512;
    bf16x16 a0 = load_a_frag(A, K, m0,      kk, lane);
    bf16x16 a1 = load_a_frag(A, K, m0 + 16, kk, lane);
    bf16x16 b0 = *(const bf16x16*)(bb0 + ko);
    bf16x16 b1 = *(const bf16x16*)(bb1 + ko);
    bf16x16 b2 = *(const bf16x16*)(bb2 + ko);
    bf16x16 b3 = *(const bf16x16*)(bb3 + ko);
    acc[0][0] = __builtin_amdgcn_wmma_f32_16x16x32_bf16(false, a0, false, b0, (short)0, acc[0][0], false, false);
    acc[1][0] = __builtin_amdgcn_wmma_f32_16x16x32_bf16(false, a1, false, b0, (short)0, acc[1][0], false, false);
    acc[0][1] = __builtin_amdgcn_wmma_f32_16x16x32_bf16(false, a0, false, b1, (short)0, acc[0][1], false, false);
    acc[1][1] = __builtin_amdgcn_wmma_f32_16x16x32_bf16(false, a1, false, b1, (short)0, acc[1][1], false, false);
    acc[0][2] = __builtin_amdgcn_wmma_f32_16x16x32_bf16(false, a0, false, b2, (short)0, acc[0][2], false, false);
    acc[1][2] = __builtin_amdgcn_wmma_f32_16x16x32_bf16(false, a1, false, b2, (short)0, acc[1][2], false, false);
    acc[0][3] = __builtin_amdgcn_wmma_f32_16x16x32_bf16(false, a0, false, b3, (short)0, acc[0][3], false, false);
    acc[1][3] = __builtin_amdgcn_wmma_f32_16x16x32_bf16(false, a1, false, b3, (short)0, acc[1][3], false, false);
    if (kk + 128 < K){
      __builtin_prefetch(bb0 + ko + 4*512, 0, 1);
      __builtin_prefetch(bb1 + ko + 4*512, 0, 1);
      __builtin_prefetch(bb2 + ko + 4*512, 0, 1);
      __builtin_prefetch(bb3 + ko + 4*512, 0, 1);
    }
  }
  const int nc = lane & 15;
  const int mb = (lane < 16) ? 0 : 8;
  #pragma unroll
  for (int mi = 0; mi < 2; mi++){
    #pragma unroll
    for (int i = 0; i < 4; i++){
      int n = n0 + i*16 + nc;
      float bn = bias[n];
      #pragma unroll
      for (int r = 0; r < 8; r++){
        int m = m0 + mi*16 + mb + r;
        float v = acc[mi][i][r] + bn;
        if (act) v = geluf(v);
        C[(size_t)m*Nlog + n] = v;
      }
    }
  }
}

// ---- small WMMA GEMM (M=64-class, ragged N): 64x64 tile, wave = 1m x 2n ----
__global__ __launch_bounds__(256)
void gemm_bf16_kernel(const float* __restrict__ A, const bf16_t* __restrict__ Bp,
                      const float* __restrict__ bias, float* __restrict__ C,
                      int M, int Nlog, int K, int act){
  const int tid = threadIdx.x;
  const int w = tid >> 5, lane = tid & 31;
  const int mt = w & 3;
  const int ntb = (w >> 2) << 1;
  const int m0 = blockIdx.x*64 + mt*16;
  const int n0a = blockIdx.y*64 + ntb*16;
  const int n0b = n0a + 16;
  const int nTper = K >> 5;
  const bf16_t* b0base = Bp + ((size_t)(n0a >> 4)*nTper)*512 + lane*16;
  const bf16_t* b1base = Bp + ((size_t)(n0b >> 4)*nTper)*512 + lane*16;
  f32x8 acc0 = {}; f32x8 acc1 = {};
  #pragma unroll 4
  for (int kk = 0; kk < K; kk += 32){
    bf16x16 af  = load_a_frag(A, K, m0, kk, lane);
    bf16x16 bf0 = *(const bf16x16*)(b0base + (size_t)(kk >> 5)*512);
    bf16x16 bf1 = *(const bf16x16*)(b1base + (size_t)(kk >> 5)*512);
    acc0 = __builtin_amdgcn_wmma_f32_16x16x32_bf16(false, af, false, bf0, (short)0, acc0, false, false);
    acc1 = __builtin_amdgcn_wmma_f32_16x16x32_bf16(false, af, false, bf1, (short)0, acc1, false, false);
  }
  const int nA = n0a + (lane & 15);
  const int nB = n0b + (lane & 15);
  const int mb = m0 + ((lane < 16) ? 0 : 8);
  #pragma unroll
  for (int r = 0; r < 8; r++){
    int m = mb + r;
    if (nA < Nlog){
      float v = acc0[r] + (bias ? bias[nA] : 0.0f);
      if (act) v = geluf(v);
      C[(size_t)m*Nlog + nA] = v;
    }
    if (nB < Nlog){
      float v = acc1[r] + (bias ? bias[nB] : 0.0f);
      if (act) v = geluf(v);
      C[(size_t)m*Nlog + nB] = v;
    }
  }
}

// ---- N=1 head: stop[row] = dot(H[row,:], w) + b ----
__global__ __launch_bounds__(256)
void rowdot_kernel(const float* __restrict__ Hd, const float* __restrict__ wv,
                   const float* __restrict__ b2, float* __restrict__ outp, int K){
  int row  = blockIdx.x*8 + (threadIdx.x >> 5);
  int lane = threadIdx.x & 31;
  const float* hr = Hd + (size_t)row*K;
  float s = 0.f;
  for (int j = lane; j < K; j += 32) s += hr[j]*wv[j];
  #pragma unroll
  for (int o = 16; o; o >>= 1) s += __shfl_xor(s, o, 32);
  if (lane == 0) outp[row] = s + b2[0];
}

// ---- prep: feats = [tokens[0], tokens[num_tokens-1]] per batch ----
__global__ __launch_bounds__(256)
void prep_feats_kernel(const float* __restrict__ tokens, const int* __restrict__ num_tokens,
                       float* __restrict__ feats){
  int idx = blockIdx.x*256 + threadIdx.x;   // 64*512
  if (idx >= BATCH*512) return;
  int b = idx >> 9, e = idx & 511;
  float v;
  if (e < A_DIM) v = tokens[(size_t)b*A_DIM + e];
  else {
    int t = num_tokens[b] - 1;
    v = tokens[((size_t)t*BATCH + b)*A_DIM + (e - A_DIM)];
  }
  feats[idx] = v;
}

// ---- prep: frames_in rows (t,b): t==0 -> init_frame(st[:,0:128]), else target[t-1] ----
__global__ __launch_bounds__(256)
void prep_frames_in_kernel(const float* __restrict__ st, const float* __restrict__ tgt,
                           float* __restrict__ fi){
  const size_t total = (size_t)S_STEPS*BATCH*C_MEL;
  for (size_t idx = blockIdx.x*256ull + threadIdx.x; idx < total; idx += (size_t)gridDim.x*256){
    size_t row = idx / C_MEL; int j = (int)(idx % C_MEL);
    int t = (int)(row / BATCH), b = (int)(row % BATCH);
    fi[idx] = (t == 0) ? st[(size_t)b*897 + j]
                       : tgt[((size_t)(t-1)*BATCH + b)*C_MEL + j];
  }
}

// ---- prep: padded token memory tok_p (B,TP,A) with beg/end pads from st ----
__global__ __launch_bounds__(256)
void prep_tokp_kernel(const float* __restrict__ st, const float* __restrict__ tokens,
                      const int* __restrict__ num_tokens, float* __restrict__ tokp){
  const size_t total = (size_t)BATCH*TP*A_DIM;
  for (size_t idx = blockIdx.x*256ull + threadIdx.x; idx < total; idx += (size_t)gridDim.x*256){
    int b = (int)(idx / ((size_t)TP*A_DIM));
    size_t rem = idx % ((size_t)TP*A_DIM);
    int i = (int)(rem / A_DIM), a = (int)(rem % A_DIM);
    int nt = num_tokens[b];
    float v;
    if (i < 16)              v = st[(size_t)b*897 + 385 + a];            // beg_pad
    else if (i - 16 < nt)    v = tokens[((size_t)(i-16)*BATCH + b)*A_DIM + a];
    else if (i < nt + 32)    v = st[(size_t)b*897 + 641 + a];            // end_pad
    else                     v = 0.0f;
    tokp[idx] = v;
  }
}

// ================= persistent attention-LSTM scan =================
// 64 WGs x 256 threads. WG wg owns gate columns [wg*32, wg*32+32) and batch b=wg
// for the attention step. cum-attention + window index live in LDS (per-batch).
__global__ __launch_bounds__(256)
void attn_scan_kernel(const float* __restrict__ pre, float* __restrict__ blk,
                      const float* __restrict__ tokp, const float* __restrict__ st,
                      float* __restrict__ X, float* __restrict__ gates, float* __restrict__ q,
                      const bf16_t* __restrict__ Wlstm, const bf16_t* __restrict__ Wq,
                      const float* __restrict__ bih, const float* __restrict__ bhh,
                      const float* __restrict__ bq, const float* __restrict__ convw,
                      const float* __restrict__ convb, const float* __restrict__ attnv,
                      const int* __restrict__ num_tokens,
                      float* __restrict__ align_out, float* __restrict__ ws_out,
                      int* __restrict__ bar){
  const int wg = blockIdx.x, tid = threadIdx.x;
  const int w = tid >> 5, lane = tid & 31;
  const int b = wg;

  __shared__ float cumS[CUMLEN];
  __shared__ float scS[33];
  __shared__ float alS[33];
  __shared__ float redS[8];
  __shared__ int   sS[2];

  // per-batch init: cum window, ctx=init_ctx, h=0
  float cinit = fabsf(st[(size_t)b*897 + 128]);
  for (int i = tid; i < CUMLEN; i += 256) cumS[i] = (i < 24) ? cinit : 0.0f;
  X[(size_t)b*1280 + 512 + tid] = st[(size_t)b*897 + 129 + tid];   // ctx (A=256=blockDim)
  for (int j = tid; j < 512; j += 256) X[(size_t)b*1280 + 768 + j] = 0.0f;
  if (tid == 0) sS[0] = 0;

  // hoisted per-thread attention constants (a = tid)
  const int a = tid;
  float cw[17];
  #pragma unroll
  for (int j = 0; j < 17; j++) cw[j] = convw[a*17 + j];
  const float cb = convb[a];
  const float va = attnv[a];
  const float* tokb = tokp + ((size_t)b*TP)*A_DIM + a;
  const int ntok = num_tokens[b];
  const int lim = ntok + 32;
  const int maxws = ntok - 1;

  // cell state lives in registers: this thread always owns the same 2 cells
  float creg0 = 0.0f, creg1 = 0.0f;

  gsync(bar);

  for (int t = 0; t < S_STEPS; ++t){
    // ---- A: X[:,0:512] = pre[t] ----
    {
      int idx0 = wg*256 + tid;
      #pragma unroll
      for (int it = 0; it < 2; ++it){
        int idx = idx0 + it*16384;
        int bb = idx >> 9, j = idx & 511;
        X[(size_t)bb*1280 + j] = pre[(((size_t)t*BATCH) + bb)*H_DIM + j];
      }
    }
    gsync(bar);
    // ---- B: gates = [x,ctx,h] @ W^T + (bih+bhh), this WG's 32 cols ----
    {
      int mt = w & 3, nt = w >> 2;
      int m0 = mt*16;
      int n0 = wg*32 + nt*16;
      f32x8 acc = {};
      const bf16_t* bbase = Wlstm + ((size_t)(n0 >> 4)*40)*512 + lane*16;
      #pragma unroll 4
      for (int kk = 0; kk < 1280; kk += 32){
        bf16x16 af = load_a_frag(X, 1280, m0, kk, lane);
        bf16x16 bf = *(const bf16x16*)(bbase + (size_t)(kk >> 5)*512);
        acc = __builtin_amdgcn_wmma_f32_16x16x32_bf16(false, af, false, bf, (short)0, acc, false, false);
      }
      int n  = n0 + (lane & 15);
      int mb = m0 + ((lane < 16) ? 0 : 8);
      float bsum = bih[n] + bhh[n];
      #pragma unroll
      for (int r = 0; r < 8; r++) gates[(size_t)(mb + r)*2048 + n] = acc[r] + bsum;
    }
    gsync(bar);
    // ---- C: LSTM cell update; write h into X; fr into blk ----
    {
      int idx0 = wg*256 + tid;
      #pragma unroll
      for (int it = 0; it < 2; ++it){
        int idx = idx0 + it*16384;
        int bb = idx >> 9, j = idx & 511;
        const float* gr = gates + (size_t)bb*2048 + j;
        float ig = sigf(gr[0]);
        float fg = sigf(gr[512]);
        float gg = tanhf(gr[1024]);
        float og = sigf(gr[1536]);
        float cprev = it ? creg1 : creg0;
        float cnew  = fg*cprev + ig*gg;
        if (it) creg1 = cnew; else creg0 = cnew;
        float hh = og*tanhf(cnew);
        X[(size_t)bb*1280 + 768 + j] = hh;
        float pv = pre[(((size_t)t*BATCH) + bb)*H_DIM + j];
        blk[(((size_t)t*BATCH) + bb)*768 + j] = (pv + hh)*INV_SQRT2;
      }
    }
    gsync(bar);
    // ---- D: q = h @ Wq^T + bq (WGs 0..7, 64 wave-tiles) ----
    if (wg < 8){
      int id = wg*8 + w;
      int m0 = (id & 3)*16, n0 = (id >> 2)*16;
      f32x8 acc = {};
      const bf16_t* bbase = Wq + ((size_t)(n0 >> 4)*16)*512 + lane*16;
      #pragma unroll 4
      for (int kk = 0; kk < 512; kk += 32){
        bf16x16 af = load_a_frag(X + 768, 1280, m0, kk, lane);
        bf16x16 bf = *(const bf16x16*)(bbase + (size_t)(kk >> 5)*512);
        acc = __builtin_amdgcn_wmma_f32_16x16x32_bf16(false, af, false, bf, (short)0, acc, false, false);
      }
      int n  = n0 + (lane & 15);
      int mb = m0 + ((lane < 16) ? 0 : 8);
      float bn = bq[n];
      #pragma unroll
      for (int r = 0; r < 8; r++) q[(mb + r)*A_DIM + n] = acc[r] + bn;
    }
    gsync(bar);
    // ---- E: location-sensitive attention for batch b ----
    {
      int s0 = sS[0];
      float qa = q[b*A_DIM + a];
      #pragma unroll 1
      for (int x = 0; x < 33; x++){
        float loc = cb;
        #pragma unroll
        for (int j = 0; j < 17; j++) loc += cumS[s0 + x + j]*cw[j];
        float v = tanhf(qa + tokb[(size_t)(s0 + x)*A_DIM] + loc)*va;
        #pragma unroll
        for (int o = 16; o; o >>= 1) v += __shfl_xor(v, o, 32);
        if (lane == 0) redS[w] = v;
        __syncthreads();
        if (tid == 0){
          float s = 0.f;
          #pragma unroll
          for (int k = 0; k < 8; k++) s += redS[k];
          scS[x] = (s0 + x < lim) ? s : -1.0e9f;
        }
        __syncthreads();
      }
      if (tid == 0){
        float mx = scS[0]; int am = 0;
        for (int x = 1; x < 33; x++) if (scS[x] > mx){ mx = scS[x]; am = x; }
        float ss = 0.f;
        for (int x = 0; x < 33; x++){ float e = __expf(scS[x] - mx); alS[x] = e; ss += e; }
        float inv = 1.0f/ss;
        for (int x = 0; x < 33; x++) alS[x] *= inv;
        int ns = s0 + am - 16;
        ns = ns < 0 ? 0 : (ns > maxws ? maxws : ns);
        sS[1] = ns;
        ws_out[(size_t)t*BATCH + b] = (float)ns;
      }
      __syncthreads();
      float ctxa = 0.f;
      #pragma unroll 1
      for (int x = 0; x < 33; x++) ctxa += alS[x]*tokb[(size_t)(s0 + x)*A_DIM];
      X[(size_t)b*1280 + 512 + a] = ctxa;
      blk[(((size_t)t*BATCH) + b)*768 + 512 + a] = ctxa;
      if (tid < 33){
        cumS[s0 + 8 + tid] += alS[tid];
        align_out[(((size_t)t*BATCH) + b)*TP + s0 + tid] = alS[tid];
      }
      __syncthreads();
      if (tid == 0) sS[0] = sS[1];
    }
    gsync(bar);
  }
}

// ================= persistent output-LSTM scan =================
__global__ __launch_bounds__(256)
void olstm_scan_kernel(float* __restrict__ pre, const float* __restrict__ blk,
                       float* __restrict__ X, float* __restrict__ gates,
                       const bf16_t* __restrict__ W, const float* __restrict__ bih,
                       const float* __restrict__ bhh, int* __restrict__ bar){
  const int wg = blockIdx.x, tid = threadIdx.x;
  const int w = tid >> 5, lane = tid & 31;

  for (int j = tid; j < 512; j += 256) X[(size_t)wg*1280 + 768 + j] = 0.0f;
  float creg0 = 0.0f, creg1 = 0.0f;
  gsync(bar);

  for (int t = 0; t < S_STEPS; ++t){
    // A: X[:,0:768] = blk[t]
    {
      int idx0 = wg*256 + tid;
      #pragma unroll
      for (int it = 0; it < 3; ++it){
        int idx = idx0 + it*16384;
        int bb = idx/768, j = idx%768;
        X[(size_t)bb*1280 + j] = blk[(((size_t)t*BATCH) + bb)*768 + j];
      }
    }
    gsync(bar);
    // B: gates
    {
      int mt = w & 3, nt = w >> 2;
      int m0 = mt*16;
      int n0 = wg*32 + nt*16;
      f32x8 acc = {};
      const bf16_t* bbase = W + ((size_t)(n0 >> 4)*40)*512 + lane*16;
      #pragma unroll 4
      for (int kk = 0; kk < 1280; kk += 32){
        bf16x16 af = load_a_frag(X, 1280, m0, kk, lane);
        bf16x16 bf = *(const bf16x16*)(bbase + (size_t)(kk >> 5)*512);
        acc = __builtin_amdgcn_wmma_f32_16x16x32_bf16(false, af, false, bf, (short)0, acc, false, false);
      }
      int n  = n0 + (lane & 15);
      int mb = m0 + ((lane < 16) ? 0 : 8);
      float bsum = bih[n] + bhh[n];
      #pragma unroll
      for (int r = 0; r < 8; r++) gates[(size_t)(mb + r)*2048 + n] = acc[r] + bsum;
    }
    gsync(bar);
    // C: cell update; fr2 = (pre + h)/sqrt2 in-place into pre
    {
      int idx0 = wg*256 + tid;
      #pragma unroll
      for (int it = 0; it < 2; ++it){
        int idx = idx0 + it*16384;
        int bb = idx >> 9, j = idx & 511;
        const float* gr = gates + (size_t)bb*2048 + j;
        float ig = sigf(gr[0]);
        float fg = sigf(gr[512]);
        float gg = tanhf(gr[1024]);
        float og = sigf(gr[1536]);
        float cprev = it ? creg1 : creg0;
        float cnew  = fg*cprev + ig*gg;
        if (it) creg1 = cnew; else creg0 = cnew;
        float hh = og*tanhf(cnew);
        X[(size_t)bb*1280 + 768 + j] = hh;
        size_t pi = (((size_t)t*BATCH) + bb)*H_DIM + j;
        pre[pi] = (pre[pi] + hh)*INV_SQRT2;
      }
    }
    gsync(bar);
  }
}

// =================== host launch ===================
extern "C" void kernel_launch(void* const* d_in, const int* in_sizes, int n_in,
                              void* d_out, int out_size, void* d_ws, size_t ws_size,
                              hipStream_t stream){
  (void)in_sizes; (void)n_in; (void)out_size; (void)ws_size;
  const float* tokens        = (const float*)d_in[0];
  const int*   num_tokens    = (const int*)d_in[2];
  const float* target_frames = (const float*)d_in[3];
  const float* init_W1 = (const float*)d_in[4];
  const float* init_b1 = (const float*)d_in[5];
  const float* init_W2 = (const float*)d_in[6];
  const float* init_b2 = (const float*)d_in[7];
  const float* pre_W1  = (const float*)d_in[8];
  const float* pre_b1  = (const float*)d_in[9];
  const float* pre_W2  = (const float*)d_in[10];
  const float* pre_b2  = (const float*)d_in[11];
  const float* lstm_Wih = (const float*)d_in[12];
  const float* lstm_Whh = (const float*)d_in[13];
  const float* lstm_bih = (const float*)d_in[14];
  const float* lstm_bhh = (const float*)d_in[15];
  const float* attn_Wq  = (const float*)d_in[16];
  const float* attn_bq  = (const float*)d_in[17];
  const float* attn_conv_w = (const float*)d_in[18];
  const float* attn_conv_b = (const float*)d_in[19];
  const float* attn_v   = (const float*)d_in[20];
  const float* stop_W1  = (const float*)d_in[21];
  const float* stop_b1  = (const float*)d_in[22];
  const float* stop_W2  = (const float*)d_in[23];
  const float* stop_b2  = (const float*)d_in[24];
  const float* olstm_Wih = (const float*)d_in[25];
  const float* olstm_Whh = (const float*)d_in[26];
  const float* olstm_bih = (const float*)d_in[27];
  const float* olstm_bhh = (const float*)d_in[28];
  const float* out_W1 = (const float*)d_in[29];
  const float* out_b1 = (const float*)d_in[30];
  const float* out_W2 = (const float*)d_in[31];
  const float* out_b2 = (const float*)d_in[32];

  float* frames_out = (float*)d_out;
  float* stop_out   = frames_out + (size_t)S_STEPS*BATCH*C_MEL;
  float* align_out  = stop_out   + (size_t)S_STEPS*BATCH;
  float* ws_out     = align_out  + (size_t)S_STEPS*BATCH*TP;

  char* wp = (char*)d_ws;
  auto alloc = [&](size_t bytes)->char*{
    char* r = wp; wp += (bytes + 255) & ~(size_t)255; return r;
  };
  // packed bf16 weights (tiles * 512 elems * 2B)
  bf16_t* p_init1 = (bf16_t*)alloc((size_t)256 *512*2);
  bf16_t* p_init2 = (bf16_t*)alloc((size_t)480 *512*2);
  bf16_t* p_pre1  = (bf16_t*)alloc((size_t)128 *512*2);
  bf16_t* p_pre2  = (bf16_t*)alloc((size_t)512 *512*2);
  bf16_t* p_lstm  = (bf16_t*)alloc((size_t)5120*512*2);
  bf16_t* p_wq    = (bf16_t*)alloc((size_t)256 *512*2);
  bf16_t* p_stop1 = (bf16_t*)alloc((size_t)768 *512*2);
  bf16_t* p_olstm = (bf16_t*)alloc((size_t)5120*512*2);
  bf16_t* p_out1  = (bf16_t*)alloc((size_t)512 *512*2);
  bf16_t* p_out2  = (bf16_t*)alloc((size_t)128 *512*2);
  // fp32 activations / state
  float* feats     = (float*)alloc((size_t)BATCH*512*4);
  float* sth       = (float*)alloc((size_t)BATCH*256*4);
  float* st        = (float*)alloc((size_t)BATCH*897*4);
  float* frames_in = (float*)alloc((size_t)S_STEPS*BATCH*C_MEL*4);
  float* pre       = (float*)alloc((size_t)S_STEPS*BATCH*H_DIM*4);
  float* tmp       = (float*)alloc((size_t)S_STEPS*BATCH*H_DIM*4);
  float* blk       = (float*)alloc((size_t)S_STEPS*BATCH*768*4);
  float* tokp      = (float*)alloc((size_t)BATCH*TP*A_DIM*4);
  float* X         = (float*)alloc((size_t)BATCH*1280*4);
  float* gates     = (float*)alloc((size_t)BATCH*2048*4);
  float* q         = (float*)alloc((size_t)BATCH*A_DIM*4);
  int*   bar       = (int*)  alloc(256);

  hipMemsetAsync(align_out, 0, (size_t)S_STEPS*BATCH*TP*4, stream);

  // weight packing
  pack_b_kernel<<<256, 256, 0, stream>>>(init_W1, p_init1, 256, 512, 512, 0);
  pack_b_kernel<<<480, 256, 0, stream>>>(init_W2, p_init2, 897, 256, 256, 0);
  pack_b_kernel<<<128, 256, 0, stream>>>(pre_W1,  p_pre1,  512, 128, 128, 0);
  pack_b_kernel<<<512, 256, 0, stream>>>(pre_W2,  p_pre2,  512, 512, 512, 0);
  pack_b_kernel<<<3072,256, 0, stream>>>(lstm_Wih, p_lstm, 2048, 768, 1280, 0);
  pack_b_kernel<<<2048,256, 0, stream>>>(lstm_Whh, p_lstm, 2048, 512, 1280, 768);
  pack_b_kernel<<<256, 256, 0, stream>>>(attn_Wq, p_wq,   256, 512, 512, 0);
  pack_b_kernel<<<768, 256, 0, stream>>>(stop_W1, p_stop1, 512, 768, 768, 0);
  pack_b_kernel<<<3072,256, 0, stream>>>(olstm_Wih, p_olstm, 2048, 768, 1280, 0);
  pack_b_kernel<<<2048,256, 0, stream>>>(olstm_Whh, p_olstm, 2048, 512, 1280, 768);
  pack_b_kernel<<<512, 256, 0, stream>>>(out_W1, p_out1, 512, 512, 512, 0);
  pack_b_kernel<<<128, 256, 0, stream>>>(out_W2, p_out2, 128, 512, 512, 0);

  // init head: st = gelu(feats@W1^T+b1)@W2^T+b2  (small M=64 path)
  prep_feats_kernel<<<128, 256, 0, stream>>>(tokens, num_tokens, feats);
  gemm_bf16_kernel<<<dim3(1, 4),  256, 0, stream>>>(feats, p_init1, init_b1, sth, 64, 256, 512, 1);
  gemm_bf16_kernel<<<dim3(1, 15), 256, 0, stream>>>(sth, p_init2, init_b2, st, 64, 897, 256, 0);

  // prep token memory / prenet input
  prep_frames_in_kernel<<<2048, 256, 0, stream>>>(st, target_frames, frames_in);
  prep_tokp_kernel<<<4096, 256, 0, stream>>>(st, tokens, num_tokens, tokp);

  // prenet (two gelu GEMMs, big path: 128x128 tiles)
  gemm_bf16_big_kernel<<<dim3(256, 4), 256, 0, stream>>>(frames_in, p_pre1, pre_b1, tmp, 512, 128, 1);
  gemm_bf16_big_kernel<<<dim3(256, 4), 256, 0, stream>>>(tmp, p_pre2, pre_b2, pre, 512, 512, 1);

  // attention LSTM persistent scan
  hipMemsetAsync(bar, 0, 8, stream);
  attn_scan_kernel<<<NWG, 256, 0, stream>>>(pre, blk, tokp, st, X, gates, q,
                                            p_lstm, p_wq, lstm_bih, lstm_bhh,
                                            attn_bq, attn_conv_w, attn_conv_b, attn_v,
                                            num_tokens, align_out, ws_out, bar);

  // stop head
  gemm_bf16_big_kernel<<<dim3(256, 4), 256, 0, stream>>>(blk, p_stop1, stop_b1, tmp, 512, 768, 1);
  rowdot_kernel<<<4096, 256, 0, stream>>>(tmp, stop_W2, stop_b2, stop_out, 512);

  // output LSTM persistent scan (writes fr2 into `pre` in place)
  hipMemsetAsync(bar, 0, 8, stream);
  olstm_scan_kernel<<<NWG, 256, 0, stream>>>(pre, blk, X, gates, p_olstm,
                                             olstm_bih, olstm_bhh, bar);

  // output head
  gemm_bf16_big_kernel<<<dim3(256, 4), 256, 0, stream>>>(pre, p_out1, out_b1, tmp, 512, 512, 1);
  gemm_bf16_big_kernel<<<dim3(256, 1), 256, 0, stream>>>(tmp, p_out2, out_b2, frames_out, 128, 512, 0);
}
